// HGT_42425686950471
// MI455X (gfx1250) — compile-verified, matrix operand order checked by reading, not run on hardware
//
#include <hip/hip_runtime.h>
#include <hip/hip_bf16.h>
#include <math.h>

// ---------------------------------------------------------------------------
// MI455X (gfx1250) HGT forward.
// Dense GEMMs: v_wmma_f32_16x16x32_bf16; weights pre-transposed to bf16 [N][K]
// so all LDS traffic is ds_load_b128/ds_store_b128; activations converted
// f32->bf16 at tile-fill time. GELU runs in a separate elementwise kernel so
// the WMMA inner loop is branch-free. Sparse attention: VALU + f32 atomics.
// ---------------------------------------------------------------------------

typedef __attribute__((ext_vector_type(16))) __bf16 bf16x16;
typedef __attribute__((ext_vector_type(8)))  float  floatx8;

__device__ __forceinline__ float sigmoidf_(float x){ return 1.0f/(1.0f+__expf(-x)); }
__device__ __forceinline__ float geluf_(float x){ return 0.5f*x*(1.0f+erff(x*0.70710678118654752440f)); }

__device__ __forceinline__ void atomicMaxF(float* addr, float val){
  unsigned int* ua = (unsigned int*)addr;
  unsigned int old = __float_as_uint(*addr);
  while (__uint_as_float(old) < val) {
    unsigned int assumed = old;
    old = atomicCAS(ua, assumed, __float_as_uint(val));
    if (old == assumed) break;
  }
}

// ---------------------------------------------------------------------------
// Weight prep: Wt[n*K + k] = bf16(W[k*N + n])   (write-coalesced)
// ---------------------------------------------------------------------------
__global__ __launch_bounds__(256)
void conv_tr_bf16(const float* __restrict__ W, __bf16* __restrict__ Wt, int K, int N)
{
  int k = blockIdx.x*32 + (threadIdx.x & 31);
  int n = blockIdx.y*8  + (threadIdx.x >> 5);
  if (k < K && n < N) Wt[(long)n*K + k] = (__bf16)W[(long)k*N + n];
}

__global__ __launch_bounds__(256)
void conv_bf16(const float* __restrict__ in, __bf16* __restrict__ outp, long n)
{
  long i = (long)blockIdx.x*blockDim.x + threadIdx.x;
  if (i < n) outp[i] = (__bf16)in[i];
}

__global__ __launch_bounds__(256)
void gelu_f32(const float* __restrict__ in, float* __restrict__ outp, long n)
{
  long i = (long)blockIdx.x*blockDim.x + threadIdx.x;
  if (i < n) outp[i] = geluf_(in[i]);
}

// ---------------------------------------------------------------------------
// Generic GEMM: C = [accum +=] alpha * postRelu( A @ W + bias )
// A: [M,K] f32 row-major.  Wt: [N,K] bf16 row-major (pre-transposed weight).
// Block 256 = 8 waves; tile BM=128, BN=64, BK=32; wave w owns rows
// [w*16, w*16+16) x 64 cols = 4 WMMA tiles. Fragment layout per ISA 7.12.2:
// lane%16 = M-row (A) / N-col (B); elem x -> K = half*8 + x (+8 when x>=8).
// ---------------------------------------------------------------------------
__global__ __launch_bounds__(256)
void gemm_bias_act(const float* __restrict__ A, const __bf16* __restrict__ Wt,
                   const float* __restrict__ bias, float* __restrict__ C,
                   int M, int N, int K, int postRelu,
                   const float* __restrict__ alphap, int accum)
{
  __shared__ __bf16 As[128][40];
  __shared__ __bf16 Bs[64][40];   // [n][k]
  const int t    = threadIdx.x;
  const int wv   = t >> 5;
  const int lane = t & 31;
  const int l16  = lane & 15;
  const int half = lane >> 4;
  const long m0 = (long)blockIdx.x * 128;
  const long n0 = (long)blockIdx.y * 64;

  floatx8 acc[4] = {};

  for (int kk = 0; kk < K; kk += 32) {
    {   // A tile 128x32: 16 consecutive f32 per thread -> 4x b128 loads
      int r0 = t >> 1, c0 = (t & 1) * 16;
      long gr = m0 + r0;
      float va[16];
      if (gr < M) {
        const float* ap = A + gr*(long)K + kk + c0;
        if (kk + 32 < K) __builtin_prefetch(ap + 32, 0, 1);
        #pragma unroll
        for (int i = 0; i < 16; ++i) va[i] = ap[i];
      } else {
        #pragma unroll
        for (int i = 0; i < 16; ++i) va[i] = 0.0f;
      }
      #pragma unroll
      for (int i = 0; i < 16; ++i) As[r0][c0 + i] = (__bf16)va[i];
    }
    {   // B tile 64x32 (Wt rows): 8 consecutive bf16 per thread
      int c0 = t >> 2, r0 = (t & 3) * 8;
      const __bf16* wp = Wt + (n0 + c0)*(long)K + kk + r0;
      #pragma unroll
      for (int i = 0; i < 8; ++i) Bs[c0][r0 + i] = wp[i];
    }
    __syncthreads();

    bf16x16 af;
    #pragma unroll
    for (int x = 0; x < 16; ++x) {
      int kl = half*8 + x + ((x >> 3) << 3);
      af[x] = As[wv*16 + l16][kl];
    }
    bf16x16 bfr[4];
    #pragma unroll
    for (int j = 0; j < 4; ++j) {
      #pragma unroll
      for (int x = 0; x < 16; ++x) {
        int kl = half*8 + x + ((x >> 3) << 3);
        bfr[j][x] = Bs[j*16 + l16][kl];
      }
    }
    #pragma unroll
    for (int j = 0; j < 4; ++j)
      acc[j] = __builtin_amdgcn_wmma_f32_16x16x32_bf16(false, af, false, bfr[j],
                                                       (short)0, acc[j], false, false);
    __syncthreads();
  }

  float alpha = alphap ? *alphap : 1.0f;
  #pragma unroll
  for (int j = 0; j < 4; ++j) {
    long col = n0 + j*16 + l16;
    #pragma unroll
    for (int r = 0; r < 8; ++r) {
      long row = m0 + wv*16 + half*8 + r;   // C layout: vgpr r -> M = r (+8 hi half)
      if (row < M) {
        float v = acc[j][r];
        if (bias) v += bias[col];
        if (postRelu) v = fmaxf(v, 0.0f);
        v *= alpha;
        long o = row*(long)N + col;
        C[o] = accum ? (C[o] + v) : v;
      }
    }
  }
}

// ---------------------------------------------------------------------------
// A_head: C[M,N] = sigmoid(Zs[M,64] @ Zd[N,64]^T), both operands bf16 [*,64].
// Row-major Z is already [n][k] for the B side -> everything contiguous.
// ---------------------------------------------------------------------------
__global__ __launch_bounds__(256)
void gemm_nt_sigmoid(const __bf16* __restrict__ Zs, const __bf16* __restrict__ Zd,
                     float* __restrict__ Cout, int M, int N)
{
  __shared__ __bf16 As[128][40];
  __shared__ __bf16 Bs[64][40];
  const int t    = threadIdx.x;
  const int wv   = t >> 5;
  const int lane = t & 31;
  const int l16  = lane & 15;
  const int half = lane >> 4;
  const long m0 = (long)blockIdx.x * 128;
  const long n0 = (long)blockIdx.y * 64;

  floatx8 acc[4] = {};

  #pragma unroll
  for (int kk = 0; kk < 64; kk += 32) {
    {   // A tile 128x32: 16 consecutive bf16 per thread -> 2x b128
      int r0 = t >> 1, c0 = (t & 1) * 16;
      long gr = m0 + r0;
      #pragma unroll
      for (int i = 0; i < 16; ++i)
        As[r0][c0 + i] = (gr < M) ? Zs[gr*64 + kk + c0 + i] : (__bf16)0.0f;
    }
    {   // B tile 64x32: 8 consecutive bf16 per thread
      int c0 = t >> 2, r0 = (t & 3) * 8;
      long gc = n0 + c0;
      #pragma unroll
      for (int i = 0; i < 8; ++i)
        Bs[c0][r0 + i] = (gc < N) ? Zd[gc*64 + kk + r0 + i] : (__bf16)0.0f;
    }
    __syncthreads();

    bf16x16 af;
    #pragma unroll
    for (int x = 0; x < 16; ++x) {
      int kl = half*8 + x + ((x >> 3) << 3);
      af[x] = As[wv*16 + l16][kl];
    }
    bf16x16 bfr[4];
    #pragma unroll
    for (int j = 0; j < 4; ++j) {
      #pragma unroll
      for (int x = 0; x < 16; ++x) {
        int kl = half*8 + x + ((x >> 3) << 3);
        bfr[j][x] = Bs[j*16 + l16][kl];
      }
    }
    #pragma unroll
    for (int j = 0; j < 4; ++j)
      acc[j] = __builtin_amdgcn_wmma_f32_16x16x32_bf16(false, af, false, bfr[j],
                                                       (short)0, acc[j], false, false);
    __syncthreads();
  }

  #pragma unroll
  for (int j = 0; j < 4; ++j) {
    long col = n0 + j*16 + l16;
    #pragma unroll
    for (int r = 0; r < 8; ++r) {
      long row = m0 + wv*16 + half*8 + r;
      if (row < M && col < N) Cout[row*(long)N + col] = sigmoidf_(acc[j][r]);
    }
  }
}

// ---------------------------------------------------------------------------
// ke[n,h,:] = k[n,h,:] @ R[h]  (R: [8,32,32]); block = node, thread = (h,e)
// ---------------------------------------------------------------------------
__global__ __launch_bounds__(256)
void rel_transform(const float* __restrict__ in, const float* __restrict__ R,
                   float* __restrict__ outp, int Nn)
{
  int n = blockIdx.x;
  if (n >= Nn) return;
  int h = threadIdx.x >> 5;
  int e = threadIdx.x & 31;
  const float* ip = in + (long)n*256 + h*32;
  const float* rp = R + h*1024 + e;
  float acc = 0.0f;
  #pragma unroll
  for (int d = 0; d < 32; ++d) acc += ip[d] * rp[d*32];
  outp[(long)n*256 + h*32 + e] = acc;
}

__global__ __launch_bounds__(256)
void fill_f32(float* __restrict__ p, float v, long n)
{
  long i = (long)blockIdx.x*blockDim.x + threadIdx.x;
  if (i < n) p[i] = v;
}

// logits + segment max (thread per (edge, head))
__global__ __launch_bounds__(256)
void edge_logits(const int* __restrict__ src, const int* __restrict__ dst,
                 const float* __restrict__ q, const float* __restrict__ ke,
                 const float* __restrict__ prel, float* __restrict__ elog,
                 float* __restrict__ mbuf, int E)
{
  int idx = blockIdx.x*blockDim.x + threadIdx.x;
  if (idx >= E*8) return;
  int e = idx >> 3, h = idx & 7;
  int s = src[e], d = dst[e];
  const float* qp = q  + (long)d*256 + h*32;
  const float* kp = ke + (long)s*256 + h*32;
  float acc = 0.0f;
  #pragma unroll
  for (int i = 0; i < 32; ++i) acc += qp[i]*kp[i];
  float lg = acc * prel[h] * 0.17677669529663687f;  // 1/sqrt(DH=32)
  elog[idx] = lg;
  atomicMaxF(mbuf + (long)d*8 + h, lg);
}

// exp(logit - max) + segment sum
__global__ __launch_bounds__(256)
void edge_exp(const int* __restrict__ dst, float* __restrict__ elog,
              const float* __restrict__ mbuf, float* __restrict__ sbuf, int E)
{
  int idx = blockIdx.x*blockDim.x + threadIdx.x;
  if (idx >= E*8) return;
  int e = idx >> 3, h = idx & 7;
  int d = dst[e];
  float ex = __expf(elog[idx] - mbuf[(long)d*8 + h]);
  elog[idx] = ex;
  atomicAdd(sbuf + (long)d*8 + h, ex);
}

// agg[dst,h,:] += alpha * ve[src,h,:]  (wave per (edge,head), lane per dim)
__global__ __launch_bounds__(256)
void edge_agg(const int* __restrict__ src, const int* __restrict__ dst,
              const float* __restrict__ elog, const float* __restrict__ sbuf,
              const float* __restrict__ ve, float* __restrict__ agg, int E)
{
  int pair = blockIdx.x*8 + (threadIdx.x >> 5);
  int lane = threadIdx.x & 31;
  if (pair >= E*8) return;
  int e = pair >> 3, h = pair & 7;
  int s = src[e], d = dst[e];
  float alpha = elog[pair] / (sbuf[(long)d*8 + h] + 1e-16f);
  atomicAdd(&agg[(long)d*256 + h*32 + lane], ve[(long)s*256 + h*32 + lane] * alpha);
}

__global__ __launch_bounds__(256)
void skip_combine(float* __restrict__ xs, const float* __restrict__ o,
                  const float* __restrict__ skipv, long n)
{
  long i = (long)blockIdx.x*blockDim.x + threadIdx.x;
  if (i >= n) return;
  float a = sigmoidf_(skipv[0]);
  xs[i] = a*o[i] + (1.0f - a)*xs[i];
}

// T_head: softmax over 2 classes of (Z @ tcls_W + b) * wt
__global__ __launch_bounds__(256)
void tcls_head(const float* __restrict__ Z, const float* __restrict__ Wc,
               const float* __restrict__ bc, const float* __restrict__ wts,
               float* __restrict__ outp, int Nn)
{
  int n = blockIdx.x*blockDim.x + threadIdx.x;
  if (n >= Nn) return;
  float u0 = bc[0], u1 = bc[1];
  const float* z = Z + (long)n*64;
  #pragma unroll 8
  for (int i = 0; i < 64; ++i) { u0 += z[i]*Wc[i*2]; u1 += z[i]*Wc[i*2+1]; }
  u0 *= wts[0]; u1 *= wts[1];
  float m  = fmaxf(u0, u1);
  float e0 = __expf(u0-m), e1 = __expf(u1-m);
  float s  = e0 + e1;
  outp[(long)n*2]   = e0/s;
  outp[(long)n*2+1] = e1/s;
}

__global__ void softmax_w(const float* __restrict__ wv_in, const float* __restrict__ wt_in,
                          float* __restrict__ wvs, float* __restrict__ wts,
                          float* __restrict__ out_wv, float* __restrict__ out_wt)
{
  if (threadIdx.x == 0) {
    float m = wv_in[0];
    for (int i = 1; i < 6; ++i) m = fmaxf(m, wv_in[i]);
    float e[6], s = 0.0f;
    for (int i = 0; i < 6; ++i) { e[i] = __expf(wv_in[i]-m); s += e[i]; }
    for (int i = 0; i < 6; ++i) { float v = e[i]/s; wvs[i] = v; out_wv[i] = v; }
    float mt = fmaxf(wt_in[0], wt_in[1]);
    float t0 = __expf(wt_in[0]-mt), t1 = __expf(wt_in[1]-mt), st = t0+t1;
    wts[0] = t0/st; wts[1] = t1/st;
    out_wt[0] = t0/st; out_wt[1] = t1/st;
  }
}

// ---------------------------------------------------------------------------
// Orchestration
// ---------------------------------------------------------------------------
extern "C" void kernel_launch(void* const* d_in, const int* in_sizes, int n_in,
                              void* d_out, int out_size, void* d_ws, size_t ws_size,
                              hipStream_t stream)
{
  (void)in_sizes; (void)n_in; (void)out_size; (void)ws_size;

  const int NT[3] = {6000, 3000, 7000};
  const int ST[4] = {0, 1, 0, 2};
  const int DT[4] = {1, 0, 2, 0};
  const int EC[4] = {30000, 30000, 40000, 40000};

  const float* xin[3] = {(const float*)d_in[0], (const float*)d_in[1], (const float*)d_in[2]};
  const float* linvW = (const float*)d_in[3];
  const float* linvB = (const float*)d_in[4];
  const float* kW = (const float*)d_in[5];  const float* kB = (const float*)d_in[6];
  const float* qW = (const float*)d_in[7];  const float* qB = (const float*)d_in[8];
  const float* vW = (const float*)d_in[9];  const float* vB = (const float*)d_in[10];
  const float* aW = (const float*)d_in[11]; const float* aB = (const float*)d_in[12];
  const float* skip = (const float*)d_in[13];
  const float* arel = (const float*)d_in[14];
  const float* mrel = (const float*)d_in[15];
  const float* prel = (const float*)d_in[16];
  const float* outW = (const float*)d_in[17]; const float* outB = (const float*)d_in[18];
  const float* recW = (const float*)d_in[19]; const float* recB = (const float*)d_in[20];
  const float* tclsW = (const float*)d_in[21]; const float* tclsB = (const float*)d_in[22];
  const float* w_view = (const float*)d_in[23];
  const float* w_type = (const float*)d_in[24];
  const int* srcs[4] = {(const int*)d_in[25], (const int*)d_in[27], (const int*)d_in[29], (const int*)d_in[31]};
  const int* dsts[4] = {(const int*)d_in[26], (const int*)d_in[28], (const int*)d_in[30], (const int*)d_in[32]};

  float* out = (float*)d_out;
  const size_t oA[4] = {0ul,
                        (size_t)6000*3000,
                        (size_t)6000*3000 + (size_t)3000*6000,
                        (size_t)6000*3000 + (size_t)3000*6000 + (size_t)6000*7000};
  const size_t oXb = oA[3] + (size_t)7000*6000;                 // 120,000,000
  const size_t oX[3] = {oXb, oXb + (size_t)6000*1024, oXb + (size_t)9000*1024};
  const size_t oTb = oXb + (size_t)16000*1024;
  const size_t oT[3] = {oTb, oTb + (size_t)6000*2, oTb + (size_t)9000*2};
  const size_t oWV = oTb + (size_t)16000*2;
  const size_t oWT = oWV + 6;

  // -- workspace bump allocator (float granules, 64-float aligned) --
  float* ws = (float*)d_ws;
  size_t off = 0;
  auto take  = [&](size_t n) { float* p = ws + off; off += (n + 63) & ~(size_t)63; return p; };
  auto takeh = [&](size_t n) { __bf16* p = (__bf16*)(ws + off);
                               off += (((n + 1)/2) + 63) & ~(size_t)63; return p; };
  float* Z[3];   for (int t = 0; t < 3; ++t) Z[t]   = take((size_t)NT[t]*64);
  float* xs[3];  for (int t = 0; t < 3; ++t) xs[t]  = take((size_t)NT[t]*256);
  float* kb[3];  for (int t = 0; t < 3; ++t) kb[t]  = take((size_t)NT[t]*256);
  float* qb[3];  for (int t = 0; t < 3; ++t) qb[t]  = take((size_t)NT[t]*256);
  float* vb[3];  for (int t = 0; t < 3; ++t) vb[t]  = take((size_t)NT[t]*256);
  float* agg[3]; for (int t = 0; t < 3; ++t) agg[t] = take((size_t)NT[t]*256);
  float* keb = take((size_t)7000*256);
  float* veb = take((size_t)7000*256);
  float* mb  = take((size_t)7000*8);
  float* sb  = take((size_t)7000*8);
  float* el  = take((size_t)40000*8);
  float* wvs = take(64);
  float* wts = take(64);
  // bf16 transposed weights [N][K]
  __bf16* linvWt = takeh((size_t)18*1024*256);
  __bf16* kWt    = takeh((size_t)6*256*256);
  __bf16* qWt    = takeh((size_t)6*256*256);
  __bf16* vWt    = takeh((size_t)6*256*256);
  __bf16* aWt    = takeh((size_t)6*256*256);
  __bf16* outWt  = takeh((size_t)3*256*64);
  __bf16* recWt  = takeh((size_t)3*64*1024);
  __bf16* Zbf[3]; for (int t = 0; t < 3; ++t) Zbf[t] = takeh((size_t)NT[t]*64);

  auto fill = [&](float* p, float v, long n) {
    fill_f32<<<dim3((unsigned)((n + 255)/256)), 256, 0, stream>>>(p, v, n);
  };
  auto tr = [&](const float* W, __bf16* Wt, int K, int N) {
    conv_tr_bf16<<<dim3((unsigned)((K + 31)/32), (unsigned)((N + 7)/8)), 256, 0, stream>>>(W, Wt, K, N);
  };
  auto gemm = [&](const float* A, const __bf16* Wt, const float* b, float* C,
                  int M, int N, int K, int postR, const float* alphap, int acc) {
    dim3 g((unsigned)((M + 127)/128), (unsigned)(N/64));
    gemm_bias_act<<<g, 256, 0, stream>>>(A, Wt, b, C, M, N, K, postR, alphap, acc);
  };

  // ---- weight prep (bf16, transposed) ----
  for (int t = 0; t < 3; ++t)
    for (int i = 0; i < 6; ++i)
      tr(linvW + ((size_t)t*6 + i)*1024*256, linvWt + ((size_t)t*6 + i)*1024*256, 1024, 256);
  for (int j = 0; j < 6; ++j) {
    tr(kW + (size_t)j*256*256, kWt + (size_t)j*256*256, 256, 256);
    tr(qW + (size_t)j*256*256, qWt + (size_t)j*256*256, 256, 256);
    tr(vW + (size_t)j*256*256, vWt + (size_t)j*256*256, 256, 256);
    tr(aW + (size_t)j*256*256, aWt + (size_t)j*256*256, 256, 256);
  }
  for (int t = 0; t < 3; ++t) {
    tr(outW + (size_t)t*256*64,  outWt + (size_t)t*256*64,  256, 64);
    tr(recW + (size_t)t*64*1024, recWt + (size_t)t*64*1024, 64, 1024);
  }

  // weight softmaxes (also written to output tail)
  softmax_w<<<1, 32, 0, stream>>>(w_view, w_type, wvs, wts, out + oWV, out + oWT);

  // Z = 0
  for (int t = 0; t < 3; ++t) fill(Z[t], 0.0f, (long)NT[t]*64);

  for (int i = 0; i < 6; ++i) {
    // view projection: xs = relu(x @ linv_W[t,i] + b)
    for (int t = 0; t < 3; ++t)
      gemm(xin[t], linvWt + ((size_t)t*6 + i)*1024*256, linvB + ((size_t)t*6 + i)*256,
           xs[t], NT[t], 256, 1024, 1, nullptr, 0);

    for (int l = 0; l < 2; ++l) {
      for (int t = 0; t < 3; ++t) {
        size_t wo = ((size_t)l*3 + t)*256*256, bo = ((size_t)l*3 + t)*256;
        gemm(xs[t], kWt + wo, kB + bo, kb[t], NT[t], 256, 256, 0, nullptr, 0);
        gemm(xs[t], qWt + wo, qB + bo, qb[t], NT[t], 256, 256, 0, nullptr, 0);
        gemm(xs[t], vWt + wo, vB + bo, vb[t], NT[t], 256, 256, 0, nullptr, 0);
      }
      for (int t = 0; t < 3; ++t) fill(agg[t], 0.0f, (long)NT[t]*256);

      for (int e = 0; e < 4; ++e) {
        int st = ST[e], dt = DT[e], E = EC[e];
        size_t ro = ((size_t)l*4 + e)*8*32*32;
        rel_transform<<<NT[st], 256, 0, stream>>>(kb[st], arel + ro, keb, NT[st]);
        rel_transform<<<NT[st], 256, 0, stream>>>(vb[st], mrel + ro, veb, NT[st]);
        fill(mb, -3.402823466e38f, (long)NT[dt]*8);
        fill(sb, 0.0f, (long)NT[dt]*8);
        unsigned gl = (unsigned)((E*8 + 255)/256);
        edge_logits<<<gl, 256, 0, stream>>>(srcs[e], dsts[e], qb[dt], keb,
                                            prel + ((size_t)l*4 + e)*8, el, mb, E);
        edge_exp<<<gl, 256, 0, stream>>>(dsts[e], el, mb, sb, E);
        edge_agg<<<(unsigned)E, 256, 0, stream>>>(srcs[e], dsts[e], el, sb, veb, agg[dt], E);
      }

      for (int t = 0; t < 3; ++t) {
        size_t wo = ((size_t)l*3 + t)*256*256, bo = ((size_t)l*3 + t)*256;
        long nel = (long)NT[t]*256;
        // g = gelu(agg) (qb free now); o = g @ aW + ab (kb free); then skip mix
        gelu_f32<<<dim3((unsigned)((nel + 255)/256)), 256, 0, stream>>>(agg[t], qb[t], nel);
        gemm(qb[t], aWt + wo, aB + bo, kb[t], NT[t], 256, 256, 0, nullptr, 0);
        skip_combine<<<dim3((unsigned)((nel + 255)/256)), 256, 0, stream>>>(
            xs[t], kb[t], skip + ((size_t)l*3 + t), nel);
      }
    }

    // Z += wv[i] * (xs @ out_W + out_b)
    for (int t = 0; t < 3; ++t)
      gemm(xs[t], outWt + (size_t)t*256*64, outB + (size_t)t*64, Z[t],
           NT[t], 64, 256, 0, wvs + i, 1);
  }

  // X_head = Z @ rec_W + rec_b
  for (int t = 0; t < 3; ++t)
    gemm(Z[t], recWt + (size_t)t*64*1024, recB + (size_t)t*1024, out + oX[t],
         NT[t], 1024, 64, 0, nullptr, 0);

  // T_head
  for (int t = 0; t < 3; ++t)
    tcls_head<<<dim3((unsigned)((NT[t] + 255)/256)), 256, 0, stream>>>(
        Z[t], tclsW + (size_t)t*64*2, tclsB + (size_t)t*2, wts, out + oT[t], NT[t]);

  // Z -> bf16 once, then A_head = sigmoid(Z[st] @ Z[dt]^T)
  for (int t = 0; t < 3; ++t)
    conv_bf16<<<dim3((unsigned)(((long)NT[t]*64 + 255)/256)), 256, 0, stream>>>(
        Z[t], Zbf[t], (long)NT[t]*64);
  for (int e = 0; e < 4; ++e) {
    int st = ST[e], dt = DT[e];
    dim3 g((unsigned)((NT[st] + 127)/128), (unsigned)((NT[dt] + 63)/64));
    gemm_nt_sigmoid<<<g, 256, 0, stream>>>(Zbf[st], Zbf[dt], out + oA[e], NT[st], NT[dt]);
  }
}
